// MultiHeadAttention_32607391711706
// MI455X (gfx1250) — matrix operand order repaired and checked by reading, not compiled, not run on hardware
//
#include <hip/hip_runtime.h>

// ---------------------------------------------------------------------------
// Multi-head causal attention, MI455X (gfx1250), wave32, bf16 WMMA pipeline.
//   1) fp32 -> bf16 convert x;  fp32 -> bf16 transposed convert of W matrices
//   2) Q/K/V = x @ W   (bf16 WMMA GEMM, f32 accum, bf16 out)
//   3) flash-style causal attention; workgroup-cooperative K/V staging into
//      LDS via global_load_async_to_lds_b128 (ASYNCcnt double-buffering),
//      WMMA QK^T and PV
//   4) out = ctx @ Wo + bo (bf16 WMMA GEMM, f32 out)
// ---------------------------------------------------------------------------

typedef __attribute__((ext_vector_type(16))) __bf16 bf16x16;
typedef __attribute__((ext_vector_type(8)))  float  f32x8;

union FragU {
    bf16x16        v;
    unsigned short h[16];
    uint4          q[2];
};

__device__ __forceinline__ unsigned short f32_to_bf16(float f) {
    unsigned int u = __float_as_uint(f);
    u = (u + 0x7FFFu + ((u >> 16) & 1u)) >> 16;   // round-to-nearest-even
    return (unsigned short)u;
}

__device__ __forceinline__ f32x8 wmma_bf16(bf16x16 a, bf16x16 b, f32x8 c) {
    // (neg_a, A, neg_b, B, c_mod, C, reuse_a, reuse_b)
    return __builtin_amdgcn_wmma_f32_16x16x32_bf16(false, a, false, b,
                                                   (short)0, c, false, false);
}

// A fragment (16x32, row-major source, leading dim ld):
//   lane<16 : row = lane,    elems 0..7 -> k=0..7,  8..15 -> k=16..23
//   lane>=16: row = lane-16, elems 0..7 -> k=8..15, 8..15 -> k=24..31
__device__ __forceinline__ bf16x16 load_frag_a(const unsigned short* base,
                                               int ld, int lane) {
    int m  = lane & 15;
    int kb = (lane < 16) ? 0 : 8;
    const unsigned short* p = base + (size_t)m * ld + kb;
    FragU f;
    f.q[0] = *(const uint4*)(p);
    f.q[1] = *(const uint4*)(p + 16);
    return f.v;
}

// B fragment (32x16) from a K-contiguous ("transposed", [N][K]) source:
//   lane<16 : col = lane,    elems 0..7 -> k=0..7,  8..15 -> k=16..23
//   lane>=16: col = lane-16, elems 0..7 -> k=8..15, 8..15 -> k=24..31
__device__ __forceinline__ bf16x16 load_frag_bt(const unsigned short* base,
                                                int ld, int lane) {
    int n  = lane & 15;
    int kb = (lane < 16) ? 0 : 8;
    const unsigned short* p = base + (size_t)n * ld + kb;
    FragU f;
    f.q[0] = *(const uint4*)(p);
    f.q[1] = *(const uint4*)(p + 16);
    return f.v;
}

// B fragment (32x16) from a row-major [K][N] source (strided, for V).
__device__ __forceinline__ bf16x16 load_frag_b_rowmajor(const unsigned short* base,
                                                        int ld, int lane) {
    int n  = lane & 15;
    int kb = (lane < 16) ? 0 : 8;
    FragU f;
#pragma unroll
    for (int e = 0; e < 16; ++e)
        f.h[e] = base[(size_t)(kb + e + (e < 8 ? 0 : 8)) * ld + n];
    return f.v;
}

__device__ __forceinline__ float rowmax16(float v) {
    v = fmaxf(v, __shfl_xor(v, 1));
    v = fmaxf(v, __shfl_xor(v, 2));
    v = fmaxf(v, __shfl_xor(v, 4));
    v = fmaxf(v, __shfl_xor(v, 8));
    return v;
}
__device__ __forceinline__ float rowsum16(float v) {
    v += __shfl_xor(v, 1);
    v += __shfl_xor(v, 2);
    v += __shfl_xor(v, 4);
    v += __shfl_xor(v, 8);
    return v;
}

// ---------------------------------------------------------------------------
// fp32 -> bf16 elementwise convert
// ---------------------------------------------------------------------------
__global__ __launch_bounds__(256) void cvt_bf16_kernel(const float* __restrict__ s,
                                                       unsigned short* __restrict__ d,
                                                       int n) {
    int i = blockIdx.x * 256 + threadIdx.x;
    if (i < n) d[i] = f32_to_bf16(s[i]);
}

// fp32 [K][N] -> bf16 [N][K] transpose-convert (32x32 LDS tiles)
__global__ __launch_bounds__(256) void transpose_cvt_kernel(const float* __restrict__ W,
                                                            unsigned short* __restrict__ Wt,
                                                            int K, int N) {
    __shared__ unsigned short tile[32][33];
    int n0 = blockIdx.x * 32;
    int k0 = blockIdx.y * 32;
    int tx = threadIdx.x & 31;
    int ty = threadIdx.x >> 5;   // 0..7
#pragma unroll
    for (int rr = 0; rr < 32; rr += 8)
        tile[ty + rr][tx] = f32_to_bf16(W[(size_t)(k0 + ty + rr) * N + n0 + tx]);
    __syncthreads();
#pragma unroll
    for (int rr = 0; rr < 32; rr += 8)
        Wt[(size_t)(n0 + ty + rr) * K + k0 + tx] = tile[tx][ty + rr];
}

// ---------------------------------------------------------------------------
// GEMM: C[M][N] = A[M][K] (bf16, row-major) @ Bt[N][K] (bf16, K-contiguous)
//   128 threads = 4 waves, each wave owns a 64x64 tile (block tile 128x128).
//   outF != nullptr -> f32 output (+bias), else bf16 output.
// ---------------------------------------------------------------------------
__global__ __launch_bounds__(128) void gemm_bf16_kernel(
    const unsigned short* __restrict__ A,
    const unsigned short* __restrict__ Bt,
    unsigned short* __restrict__ outB,
    float* __restrict__ outF,
    const float* __restrict__ bias,
    int M, int N, int K) {

    const int lane = threadIdx.x & 31;
    const int wave = threadIdx.x >> 5;
    const int m0 = blockIdx.y * 128 + (wave >> 1) * 64;
    const int n0 = blockIdx.x * 128 + (wave & 1) * 64;

    const f32x8 kZero = {0.f, 0.f, 0.f, 0.f, 0.f, 0.f, 0.f, 0.f};
    f32x8 acc[4][4];
#pragma unroll
    for (int i = 0; i < 4; ++i)
#pragma unroll
        for (int j = 0; j < 4; ++j) acc[i][j] = kZero;

    for (int k0 = 0; k0 < K; k0 += 32) {
        if (k0 + 32 < K) {  // L2 prefetch of next K slab (near temporal)
            __builtin_prefetch(A + (size_t)m0 * K + k0 + 32, 0, 3);
            __builtin_prefetch(Bt + (size_t)n0 * K + k0 + 32, 0, 3);
        }
        bf16x16 af[4], bf[4];
#pragma unroll
        for (int i = 0; i < 4; ++i)
            af[i] = load_frag_a(A + (size_t)(m0 + i * 16) * K + k0, K, lane);
#pragma unroll
        for (int j = 0; j < 4; ++j)
            bf[j] = load_frag_bt(Bt + (size_t)(n0 + j * 16) * K + k0, K, lane);
#pragma unroll
        for (int i = 0; i < 4; ++i)
#pragma unroll
            for (int j = 0; j < 4; ++j)
                acc[i][j] = wmma_bf16(af[i], bf[j], acc[i][j]);
    }

    // C layout: lane<16 -> rows r+0, lane>=16 -> rows r+8; col = lane&15
    const int n     = lane & 15;
    const int rbase = (lane < 16) ? 0 : 8;
#pragma unroll
    for (int i = 0; i < 4; ++i)
#pragma unroll
        for (int j = 0; j < 4; ++j) {
            int col  = n0 + j * 16 + n;
            float bv = bias ? bias[col] : 0.0f;
#pragma unroll
            for (int r = 0; r < 8; ++r) {
                int   row = m0 + i * 16 + rbase + r;
                float val = acc[i][j][r] + bv;
                if (outF) outF[(size_t)row * N + col] = val;
                else      outB[(size_t)row * N + col] = f32_to_bf16(val);
            }
        }
}

// ---------------------------------------------------------------------------
// Flash-style causal attention, workgroup-cooperative.
//   One 256-thread block (8 waves) owns 128 consecutive query rows of one
//   (b,h); wave w handles q-tile rows [qblk*128 + w*16, +16).
//   Each 32-key block of K and V (32x64 bf16 = 4KB each) is staged into LDS
//   by all 256 threads with ONE global_load_async_to_lds_b128 per thread per
//   tile, double-buffered on ASYNCcnt, so all 8 waves share the DMA traffic.
// ---------------------------------------------------------------------------
__global__ __launch_bounds__(256) void attn_kernel(
    const unsigned short* __restrict__ Q,
    const unsigned short* __restrict__ K,
    const unsigned short* __restrict__ V,
    unsigned short* __restrict__ ctx,
    int H, int T, int D) {

    __shared__ unsigned short kstage[2][32][64];   // keys:  [key][dim]
    __shared__ unsigned short vstage[2][32][72];   // vals:  [key][dim], padded
    __shared__ unsigned short pbuf[8][16][40];     // per-wave P bounce, padded

    const int lane = threadIdx.x & 31;
    const int wave = threadIdx.x >> 5;

    const int qblk = blockIdx.x & (T / 128 - 1);   // 16 q-blocks per head
    const int hb   = blockIdx.x >> 4;
    const int h    = hb & (H - 1);
    const int b    = hb >> 4;
    const int qbase = qblk * 128 + wave * 16;
    const int jmax  = qblk * 128 + 127;            // uniform across block

    const size_t headOff = ((size_t)b * T) * D + (size_t)h * 64;
    const unsigned short* Qh = Q + headOff;
    const unsigned short* Kh = K + headOff;
    const unsigned short* Vh = V + headOff;

    // Cooperative staging: thread -> (row 0..31, 16B chunk 0..7) of each tile.
    const int srow = threadIdx.x >> 3;             // 0..31 (key row in tile)
    const int sc8  = (threadIdx.x & 7) * 8;        // element offset of 16B chunk

    auto issue_stage = [&](int jb, int sbuf) {
        const unsigned short* gk = Kh + (size_t)(jb + srow) * D + sc8;
        const unsigned short* gv = Vh + (size_t)(jb + srow) * D + sc8;
        unsigned lk = (unsigned)(size_t)&kstage[sbuf][srow][sc8];
        unsigned lv = (unsigned)(size_t)&vstage[sbuf][srow][sc8];
        asm volatile("global_load_async_to_lds_b128 %0, %1, off"
                     :: "v"(lk), "v"((unsigned long long)gk) : "memory");
        asm volatile("global_load_async_to_lds_b128 %0, %1, off"
                     :: "v"(lv), "v"((unsigned long long)gv) : "memory");
    };

    issue_stage(0, 0);   // prime buffer 0 with key block 0

    // Q fragments for this wave's 16-row tile (hd=64 -> two 16x32 A frags)
    const bf16x16 aq0 = load_frag_a(Qh + (size_t)qbase * D + 0,  D, lane);
    const bf16x16 aq1 = load_frag_a(Qh + (size_t)qbase * D + 32, D, lane);

    const f32x8 kZero = {0.f, 0.f, 0.f, 0.f, 0.f, 0.f, 0.f, 0.f};
    f32x8 o[4];
#pragma unroll
    for (int u = 0; u < 4; ++u) o[u] = kZero;

    float mrow[8], lrow[8];
#pragma unroll
    for (int r = 0; r < 8; ++r) { mrow[r] = -3.0e38f; lrow[r] = 0.0f; }

    const int   n     = lane & 15;
    const int   rbase = (lane < 16) ? 0 : 8;
    const float scale = 0.125f;                    // 1/sqrt(64)

    int buf = 0;
    for (int jb = 0; jb <= jmax; jb += 32, buf ^= 1) {
        // Issue DMA for the next key block into the other buffer, then retire
        // exactly this block's loads (async loads complete in order).
        const bool hasNext = (jb + 32 <= jmax);    // uniform
        if (hasNext) {
            issue_stage(jb + 32, buf ^ 1);
            asm volatile("s_wait_asynccnt 2" ::: "memory");
        } else {
            asm volatile("s_wait_asynccnt 0" ::: "memory");
        }
        __syncthreads();                           // publish staged K/V

        if (qbase + 15 >= jb) {                    // causal block skip
            // ---- S = Q K^T for 32 keys (two 16x16 tiles) from LDS ----
            f32x8 s[2];
#pragma unroll
            for (int t = 0; t < 2; ++t) {
                const unsigned short* kp = &kstage[buf][t * 16][0];
                bf16x16 bk0 = load_frag_bt(kp + 0,  64, lane);
                bf16x16 bk1 = load_frag_bt(kp + 32, 64, lane);
                s[t] = wmma_bf16(aq0, bk0, kZero);
                s[t] = wmma_bf16(aq1, bk1, s[t]);
            }

            // ---- scale + causal mask ----
#pragma unroll
            for (int t = 0; t < 2; ++t)
#pragma unroll
                for (int r = 0; r < 8; ++r) {
                    int   row = qbase + rbase + r;
                    int   col = jb + t * 16 + n;
                    float sv  = s[t][r] * scale;
                    s[t][r]   = (col > row) ? -3.0e38f : sv;
                }

            // ---- online softmax ----
            float corr[8];
#pragma unroll
            for (int r = 0; r < 8; ++r) {
                float v    = fmaxf(s[0][r], s[1][r]);
                float mnew = fmaxf(mrow[r], rowmax16(v));
                corr[r]    = __expf(mrow[r] - mnew);
                mrow[r]    = mnew;
            }
#pragma unroll
            for (int t = 0; t < 2; ++t)
#pragma unroll
                for (int r = 0; r < 8; ++r)
                    s[t][r] = __expf(s[t][r] - mrow[r]);
#pragma unroll
            for (int r = 0; r < 8; ++r) {
                float rs = rowsum16(s[0][r] + s[1][r]);
                lrow[r]  = lrow[r] * corr[r] + rs;
            }
#pragma unroll
            for (int u = 0; u < 4; ++u)
#pragma unroll
                for (int r = 0; r < 8; ++r) o[u][r] *= corr[r];

            // ---- P: C-layout -> A-layout via wave-private LDS ----
#pragma unroll
            for (int t = 0; t < 2; ++t)
#pragma unroll
                for (int r = 0; r < 8; ++r)
                    pbuf[wave][rbase + r][t * 16 + n] = f32_to_bf16(s[t][r]);
            asm volatile("s_wait_dscnt 0" ::: "memory");  // cross-lane LDS vis
            bf16x16 pa = load_frag_a(&pbuf[wave][0][0], 40, lane);

            // ---- O += P @ V (32-deep, 4 column tiles of hd) from LDS ----
#pragma unroll
            for (int u = 0; u < 4; ++u) {
                bf16x16 bv = load_frag_b_rowmajor(&vstage[buf][0][u * 16], 72, lane);
                o[u] = wmma_bf16(pa, bv, o[u]);
            }
        }
        __syncthreads();   // all readers done before buf is DMA-overwritten
    }

    // ---- normalize and store ctx (bf16, [B,T,D] with head offset) ----
#pragma unroll
    for (int r = 0; r < 8; ++r) lrow[r] = 1.0f / fmaxf(lrow[r], 1e-20f);
#pragma unroll
    for (int u = 0; u < 4; ++u)
#pragma unroll
        for (int r = 0; r < 8; ++r) {
            size_t row = (size_t)b * T + qbase + rbase + r;
            ctx[row * D + h * 64 + u * 16 + n] = f32_to_bf16(o[u][r] * lrow[r]);
        }
}

// ---------------------------------------------------------------------------
// Host-side launcher
// ---------------------------------------------------------------------------
extern "C" void kernel_launch(void* const* d_in, const int* in_sizes, int n_in,
                              void* d_out, int out_size, void* d_ws, size_t ws_size,
                              hipStream_t stream) {
    (void)in_sizes; (void)n_in; (void)out_size; (void)ws_size;

    const float* x  = (const float*)d_in[0];
    const float* Wq = (const float*)d_in[1];
    const float* Wk = (const float*)d_in[2];
    const float* Wv = (const float*)d_in[3];
    const float* Wo = (const float*)d_in[4];
    const float* bo = (const float*)d_in[5];

    const int Bb = 2, T = 2048, D = 1024, H = 16;
    const int MT = Bb * T;                 // 4096 rows
    const size_t MK = (size_t)MT * D;      // 4M elements
    const size_t KK = (size_t)D * D;       // 1M elements

    unsigned short* ws = (unsigned short*)d_ws;
    size_t off = 0;
    unsigned short* xb   = ws + off; off += MK;
    unsigned short* Wqt  = ws + off; off += KK;
    unsigned short* Wkt  = ws + off; off += KK;
    unsigned short* Wvt  = ws + off; off += KK;
    unsigned short* Wot  = ws + off; off += KK;
    unsigned short* Qb   = ws + off; off += MK;
    unsigned short* Kb   = ws + off; off += MK;
    unsigned short* Vb   = ws + off; off += MK;
    unsigned short* ctxb = ws + off; off += MK;

    // 1) convert activations, transpose-convert weights
    cvt_bf16_kernel<<<(int)((MK + 255) / 256), 256, 0, stream>>>(x, xb, (int)MK);
    dim3 tgrid(D / 32, D / 32);
    transpose_cvt_kernel<<<tgrid, 256, 0, stream>>>(Wq, Wqt, D, D);
    transpose_cvt_kernel<<<tgrid, 256, 0, stream>>>(Wk, Wkt, D, D);
    transpose_cvt_kernel<<<tgrid, 256, 0, stream>>>(Wv, Wvt, D, D);
    transpose_cvt_kernel<<<tgrid, 256, 0, stream>>>(Wo, Wot, D, D);

    // 2) QKV projections
    dim3 ggrid(D / 128, MT / 128);
    gemm_bf16_kernel<<<ggrid, 128, 0, stream>>>(xb, Wqt, Qb, nullptr, nullptr, MT, D, D);
    gemm_bf16_kernel<<<ggrid, 128, 0, stream>>>(xb, Wkt, Kb, nullptr, nullptr, MT, D, D);
    gemm_bf16_kernel<<<ggrid, 128, 0, stream>>>(xb, Wvt, Vb, nullptr, nullptr, MT, D, D);

    // 3) causal attention: one block per 128-query stripe of one (b,h)
    int nblocks = Bb * H * (T / 128);      // 512
    attn_kernel<<<nblocks, 256, 0, stream>>>(Qb, Kb, Vb, ctxb, H, T, D);

    // 4) output projection + bias, f32 out
    gemm_bf16_kernel<<<ggrid, 128, 0, stream>>>(ctxb, Wot, nullptr, (float*)d_out, bo,
                                                MT, D, D);
}
